// LinearAttention_49143015801133
// MI455X (gfx1250) — matrix operand order repaired and checked by reading, not compiled, not run on hardware
//
#include <hip/hip_runtime.h>
#include <hip/hip_bf16.h>

// ---------------------------------------------------------------------------
// Linear attention (chunked / flash-linear form) for MI455X (gfx1250).
// Matrix work through v_wmma_f32_16x16x32_bf16; GEMM tiles staged to LDS via
// the gfx1250 async-to-LDS path (ASYNCcnt) with double buffering.
// ---------------------------------------------------------------------------

typedef __bf16 bf16_t;
typedef __attribute__((ext_vector_type(16))) __bf16 v16bf;
typedef __attribute__((ext_vector_type(8)))  float  v8f;
typedef __attribute__((ext_vector_type(4)))  int    v4i;

#define B_     2
#define T_     4096
#define E_     1024
#define H_     16
#define D_     64
#define CHUNK  64
#define NCHUNK (T_ / CHUNK)   // 64
#define BH_    (B_ * H_)      // 32
#define EPS_   1e-6f
#define KSTEP  64

#if defined(__gfx1250__) && __has_builtin(__builtin_amdgcn_global_load_async_to_lds_b128)
#define ASYNC_LDS_OK 1
#endif

typedef __attribute__((address_space(1))) v4i glb_v4i;   // global int4
typedef __attribute__((address_space(3))) v4i lds_v4i;   // LDS int4

// 16B global -> LDS copy; async DMA when available, sync fallback otherwise.
static __device__ __forceinline__ void copy_b128(const bf16_t* g, bf16_t* l) {
#if defined(ASYNC_LDS_OK)
    __builtin_amdgcn_global_load_async_to_lds_b128(
        (glb_v4i*)(uintptr_t)g,
        (lds_v4i*)(unsigned int)(uintptr_t)l,   // flat shared addr low 32b == LDS offset
        0, 0);
#else
    *(uint4*)l = *(const uint4*)g;
#endif
}
static __device__ __forceinline__ void wait_async_copies() {
#if defined(ASYNC_LDS_OK)
#if __has_builtin(__builtin_amdgcn_s_wait_asynccnt)
    __builtin_amdgcn_s_wait_asynccnt(0);
#else
    asm volatile("s_wait_asynccnt 0x0" ::: "memory");
#endif
#endif
}

// --- WMMA fragment loaders (wave32 layouts per CDNA5 ISA 7.12.2) ------------
static __device__ __forceinline__ v16bf ld_a_frag(const bf16_t* lo, const bf16_t* hi) {
    union { v16bf v; uint4 q[2]; } u;
    u.q[0] = *(const uint4*)lo;
    u.q[1] = *(const uint4*)hi;
    return u.v;
}
static __device__ __forceinline__ v16bf ld_b_frag(const bf16_t* lo, const bf16_t* hi) {
    union { v16bf v; uint4 q[2]; } u;
    u.q[0] = *(const uint4*)lo;
    u.q[1] = *(const uint4*)hi;
    return u.v;
}
static __device__ __forceinline__ v8f wmma_bf16(v16bf a, v16bf b, v8f c) {
    return __builtin_amdgcn_wmma_f32_16x16x32_bf16(false, a, false, b,
                                                   (short)0, c, false, false);
}
static __device__ __forceinline__ v8f v8f_zero() {
    v8f z = {0.f,0.f,0.f,0.f,0.f,0.f,0.f,0.f};
    return z;
}

// ---------------------------------------------------------------------------
// 1a) f32 -> bf16 elementwise (activations)
// ---------------------------------------------------------------------------
__global__ void f32_to_bf16_kernel(const float* __restrict__ src,
                                   bf16_t* __restrict__ dst, int n) {
    int i = blockIdx.x * blockDim.x + threadIdx.x;
    if (i < n) dst[i] = (bf16_t)src[i];
}

// ---------------------------------------------------------------------------
// 1b) Weight transpose+convert: W f32 [k][n] -> WT bf16 [n][k].
//     Done once so GEMM B-tiles are linear copies (async-to-LDS eligible).
// ---------------------------------------------------------------------------
__global__ __launch_bounds__(256)
void w_transpose_bf16_kernel(const float* __restrict__ W, bf16_t* __restrict__ WT) {
    __shared__ float tile[32][33];
    const int bx = blockIdx.x * 32;   // n base
    const int by = blockIdx.y * 32;   // k base
    const int tx = threadIdx.x & 31;
    const int ty = threadIdx.x >> 5;  // 0..7
    #pragma unroll
    for (int j = 0; j < 32; j += 8)
        tile[ty + j][tx] = W[(size_t)(by + ty + j) * E_ + bx + tx];
    __syncthreads();
    #pragma unroll
    for (int j = 0; j < 32; j += 8)
        WT[(size_t)(bx + ty + j) * E_ + by + tx] = (bf16_t)tile[tx][ty + j];
}

// ---------------------------------------------------------------------------
// 2) Tiled WMMA GEMM:  Y = act(A[M,1024] @ W + bias), W given transposed [n][k].
//    Block 256 = 8 waves; tile 128x128; wave 32x64; K-step 64; LDS double
//    buffered, next tile prefetched via async-to-LDS while WMMAs run.
//    mode 0: bf16 out to [B,H,T,D] (+ optional elu(x)+1); mode 1: f32 [M,N].
// ---------------------------------------------------------------------------
__global__ __launch_bounds__(256)
void proj_gemm_kernel(const bf16_t* __restrict__ A,    // [M,1024] row-major
                      const bf16_t* __restrict__ WT,   // [1024(n),1024(k)]
                      const float*  __restrict__ bias,
                      bf16_t* __restrict__ out_bhtd,
                      float*  __restrict__ out_flat,
                      int mode, int elu)
{
    __shared__ bf16_t As[2][128][KSTEP];   // [m][k]
    __shared__ bf16_t Bs[2][128][KSTEP];   // [n][k]

    const int tid   = threadIdx.x;
    const int nbase = blockIdx.x * 128;
    const int mbase = blockIdx.y * 128;

    const int wave = tid >> 5;
    const int lane = tid & 31;
    const int half = lane >> 4;
    const int mr   = lane & 15;
    const int wm   = wave & 3;   // m-strip 0..3 (32 rows)
    const int wn   = wave >> 2;  // n-strip 0..1 (64 cols)

    // loader role: each thread owns one contiguous 64B chunk per tile
    const int crow = tid >> 1;            // 0..127
    const int ccol = (tid & 1) * 32;      // 0 / 32 halfs
    const bf16_t* gA = A  + (size_t)(mbase + crow) * 1024 + ccol;
    const bf16_t* gB = WT + (size_t)(nbase + crow) * 1024 + ccol;

    v8f acc[2][4];
    #pragma unroll
    for (int mi = 0; mi < 2; mi++)
        #pragma unroll
        for (int nj = 0; nj < 4; nj++) acc[mi][nj] = v8f_zero();

    // prologue: stage tile kt=0 into buffer 0
    #pragma unroll
    for (int j = 0; j < 4; j++) {
        copy_b128(gA + j * 8, &As[0][crow][ccol] + j * 8);
        copy_b128(gB + j * 8, &Bs[0][crow][ccol] + j * 8);
    }
    wait_async_copies();
    __syncthreads();

    int buf = 0;
    for (int kt = 0; kt < 1024; kt += KSTEP) {
        if (kt + KSTEP < 1024) {   // prefetch next tile into other buffer
            #pragma unroll
            for (int j = 0; j < 4; j++) {
                copy_b128(gA + kt + KSTEP + j * 8, &As[buf ^ 1][crow][ccol] + j * 8);
                copy_b128(gB + kt + KSTEP + j * 8, &Bs[buf ^ 1][crow][ccol] + j * 8);
            }
        }
        #pragma unroll
        for (int kk = 0; kk < KSTEP; kk += 32) {
            v16bf afrag[2], bfrag[4];
            #pragma unroll
            for (int mi = 0; mi < 2; mi++) {
                int row = 32 * wm + 16 * mi + mr;
                afrag[mi] = ld_a_frag(&As[buf][row][kk + 8 * half],
                                      &As[buf][row][kk + 16 + 8 * half]);
            }
            #pragma unroll
            for (int nj = 0; nj < 4; nj++) {
                int col = 64 * wn + 16 * nj + mr;
                bfrag[nj] = ld_b_frag(&Bs[buf][col][kk + 8 * half], &Bs[buf][col][kk + 16 + 8 * half]);
            }
            #pragma unroll
            for (int mi = 0; mi < 2; mi++)
                #pragma unroll
                for (int nj = 0; nj < 4; nj++)
                    acc[mi][nj] = wmma_bf16(afrag[mi], bfrag[nj], acc[mi][nj]);
        }
        wait_async_copies();
        __syncthreads();
        buf ^= 1;
    }

    // Epilogue: bias + optional elu(x)+1, scatter.
    #pragma unroll
    for (int mi = 0; mi < 2; mi++) {
        #pragma unroll
        for (int nj = 0; nj < 4; nj++) {
            int n = nbase + 64 * wn + 16 * nj + mr;
            float bval = bias[n];
            #pragma unroll
            for (int r = 0; r < 8; r++) {
                int m = mbase + 32 * wm + 16 * mi + r + 8 * half;
                float v = acc[mi][nj][r] + bval;
                if (elu) v = (v > 0.0f) ? (v + 1.0f) : __expf(v);
                if (mode == 0) {
                    int b = m >> 12, t = m & (T_ - 1);
                    int h = n >> 6,  d = n & 63;
                    out_bhtd[((size_t)(b * H_ + h) * T_ + t) * D_ + d] = (bf16_t)v;
                } else {
                    out_flat[(size_t)m * 1024 + n] = v;
                }
            }
        }
    }
}

// ---------------------------------------------------------------------------
// 3) Per-chunk state:  S_c = K_chunk^T @ V_chunk  (64x64x64),  z_c = sum_t k_t
// ---------------------------------------------------------------------------
__global__ __launch_bounds__(128)
void chunk_state_kernel(const bf16_t* __restrict__ kbuf,   // [BH,T,D]
                        const bf16_t* __restrict__ vbuf,   // [BH,T,D]
                        float* __restrict__ Sbuf,          // [BH,NCHUNK,64,64]
                        float* __restrict__ zbuf)          // [BH,NCHUNK,64]
{
    __shared__ bf16_t KT[64][64];   // [d][t]
    __shared__ bf16_t VT[64][64];   // [e][t]

    const int tid = threadIdx.x;
    const int bh  = blockIdx.x / NCHUNK;
    const int c   = blockIdx.x % NCHUNK;
    const int t0  = c * CHUNK;

    {
        int trow = tid >> 1;
        int d0   = (tid & 1) * 32;
        const bf16_t* gk = kbuf + ((size_t)bh * T_ + t0 + trow) * D_ + d0;
        const bf16_t* gv = vbuf + ((size_t)bh * T_ + t0 + trow) * D_ + d0;
        #pragma unroll
        for (int j = 0; j < 32; j++) {
            KT[d0 + j][trow] = gk[j];
            VT[d0 + j][trow] = gv[j];
        }
    }
    __syncthreads();

    const int wave = tid >> 5, lane = tid & 31;
    const int half = lane >> 4, mr = lane & 15;

    v8f acc[4];
    #pragma unroll
    for (int nj = 0; nj < 4; nj++) acc[nj] = v8f_zero();

    #pragma unroll
    for (int ks = 0; ks < 64; ks += 32) {
        int arow = 16 * wave + mr;
        v16bf a = ld_a_frag(&KT[arow][ks + 8 * half], &KT[arow][ks + 16 + 8 * half]);
        #pragma unroll
        for (int nj = 0; nj < 4; nj++) {
            v16bf bfr = ld_b_frag(&VT[16 * nj + mr][ks + 8 * half], &VT[16 * nj + mr][ks + 16 + 8 * half]);
            acc[nj] = wmma_bf16(a, bfr, acc[nj]);
        }
    }

    float* Sc = Sbuf + ((size_t)bh * NCHUNK + c) * 4096;
    #pragma unroll
    for (int nj = 0; nj < 4; nj++)
        #pragma unroll
        for (int r = 0; r < 8; r++) {
            int m = 16 * wave + r + 8 * half;
            int n = 16 * nj + mr;
            Sc[m * 64 + n] = acc[nj][r];
        }

    if (tid < 64) {
        float s = 0.f;
        for (int t = 0; t < 64; t++) s += (float)KT[tid][t];
        zbuf[((size_t)bh * NCHUNK + c) * 64 + tid] = s;
    }
}

// ---------------------------------------------------------------------------
// 4) Exclusive prefix over chunks (per (b,h)): S, z.
// ---------------------------------------------------------------------------
__global__ __launch_bounds__(256)
void prefix_kernel(float* __restrict__ Sbuf, float* __restrict__ zbuf) {
    const int bh = blockIdx.x;
    const int tid = threadIdx.x;
    float accS[16];
    #pragma unroll
    for (int i = 0; i < 16; i++) accS[i] = 0.f;
    float accz = 0.f;
    for (int c = 0; c < NCHUNK; c++) {
        float* Sc = Sbuf + ((size_t)bh * NCHUNK + c) * 4096 + tid * 16;
        #pragma unroll
        for (int i = 0; i < 16; i++) {
            float t = Sc[i]; Sc[i] = accS[i]; accS[i] += t;
        }
        if (tid < 64) {
            float* zc = zbuf + ((size_t)bh * NCHUNK + c) * 64 + tid;
            float t = *zc; *zc = accz; accz += t;
        }
    }
}

// ---------------------------------------------------------------------------
// 5) Per-chunk output:
//    A = tril(Q K^T);  O = Q @ S_prev + A @ V;  den = rowsum(A)+q.z_prev+eps
// ---------------------------------------------------------------------------
__global__ __launch_bounds__(128)
void attn_chunk_kernel(const bf16_t* __restrict__ qbuf,
                       const bf16_t* __restrict__ kbuf,
                       const bf16_t* __restrict__ vbuf,
                       const float*  __restrict__ Sbuf,
                       const float*  __restrict__ zbuf,
                       bf16_t* __restrict__ attn)   // [B,T,E]
{
    __shared__ bf16_t Qs [64][64];   // [t][d]
    __shared__ bf16_t Ks [64][64];   // [t'][d]
    __shared__ bf16_t VTs[64][64];   // [e][t']
    __shared__ bf16_t STs[64][64];   // [e][d]
    __shared__ bf16_t Am [64][64];   // masked scores
    __shared__ float  den[64];

    const int tid = threadIdx.x;
    const int bh  = blockIdx.x / NCHUNK;
    const int c   = blockIdx.x % NCHUNK;
    const int t0  = c * CHUNK;
    const int b   = bh / H_;
    const int h   = bh % H_;

    {
        int trow = tid >> 1;
        int d0   = (tid & 1) * 32;
        const bf16_t* gq = qbuf + ((size_t)bh * T_ + t0 + trow) * D_ + d0;
        const bf16_t* gk = kbuf + ((size_t)bh * T_ + t0 + trow) * D_ + d0;
        const bf16_t* gv = vbuf + ((size_t)bh * T_ + t0 + trow) * D_ + d0;
        #pragma unroll
        for (int j = 0; j < 32; j += 8) {
            *(uint4*)&Qs[trow][d0 + j] = *(const uint4*)(gq + j);
            *(uint4*)&Ks[trow][d0 + j] = *(const uint4*)(gk + j);
        }
        #pragma unroll
        for (int j = 0; j < 32; j++) VTs[d0 + j][trow] = gv[j];
        const float* gs = Sbuf + ((size_t)bh * NCHUNK + c) * 4096 + trow * 64 + d0;
        #pragma unroll
        for (int j = 0; j < 32; j++) STs[d0 + j][trow] = (bf16_t)gs[j];
    }
    __syncthreads();

    const int wave = tid >> 5, lane = tid & 31;
    const int half = lane >> 4, mr = lane & 15;
    const int arow = 16 * wave + mr;

    { // scores + causal mask
        v8f accA[4];
        #pragma unroll
        for (int nj = 0; nj < 4; nj++) accA[nj] = v8f_zero();
        #pragma unroll
        for (int ks = 0; ks < 64; ks += 32) {
            v16bf a = ld_a_frag(&Qs[arow][ks + 8 * half], &Qs[arow][ks + 16 + 8 * half]);
            #pragma unroll
            for (int nj = 0; nj < 4; nj++) {
                v16bf bfr = ld_b_frag(&Ks[16 * nj + mr][ks + 8 * half], &Ks[16 * nj + mr][ks + 16 + 8 * half]);
                accA[nj] = wmma_bf16(a, bfr, accA[nj]);
            }
        }
        #pragma unroll
        for (int nj = 0; nj < 4; nj++)
            #pragma unroll
            for (int r = 0; r < 8; r++) {
                int trow = 16 * wave + r + 8 * half;
                int tcol = 16 * nj + mr;
                float v = (trow >= tcol) ? accA[nj][r] : 0.0f;
                Am[trow][tcol] = (bf16_t)v;
            }
    }
    __syncthreads();

    if (tid < 64) { // denominator
        float rs = 0.f;
        #pragma unroll 8
        for (int j = 0; j < 64; j++) rs += (float)Am[tid][j];
        const float* zp = zbuf + ((size_t)bh * NCHUNK + c) * 64;
        float qz = 0.f;
        #pragma unroll 8
        for (int d = 0; d < 64; d++) qz += (float)Qs[tid][d] * zp[d];
        den[tid] = rs + qz + EPS_;
    }

    v8f acc[4];
    #pragma unroll
    for (int nj = 0; nj < 4; nj++) acc[nj] = v8f_zero();
    #pragma unroll
    for (int ks = 0; ks < 64; ks += 32) {   // inter-chunk: Q @ S_prev
        v16bf a = ld_a_frag(&Qs[arow][ks + 8 * half], &Qs[arow][ks + 16 + 8 * half]);
        #pragma unroll
        for (int nj = 0; nj < 4; nj++) {
            v16bf bfr = ld_b_frag(&STs[16 * nj + mr][ks + 8 * half], &STs[16 * nj + mr][ks + 16 + 8 * half]);
            acc[nj] = wmma_bf16(a, bfr, acc[nj]);
        }
    }
    #pragma unroll
    for (int ks = 0; ks < 64; ks += 32) {   // intra-chunk: A @ V
        v16bf a = ld_a_frag(&Am[arow][ks + 8 * half], &Am[arow][ks + 16 + 8 * half]);
        #pragma unroll
        for (int nj = 0; nj < 4; nj++) {
            v16bf bfr = ld_b_frag(&VTs[16 * nj + mr][ks + 8 * half], &VTs[16 * nj + mr][ks + 16 + 8 * half]);
            acc[nj] = wmma_bf16(a, bfr, acc[nj]);
        }
    }
    __syncthreads();

    #pragma unroll
    for (int nj = 0; nj < 4; nj++)
        #pragma unroll
        for (int r = 0; r < 8; r++) {
            int m = 16 * wave + r + 8 * half;
            int t = t0 + m;
            int n = h * D_ + 16 * nj + mr;
            float v = acc[nj][r] / den[m];
            attn[((size_t)b * T_ + t) * E_ + n] = (bf16_t)v;
        }
}

// ---------------------------------------------------------------------------
// Host launcher
// ---------------------------------------------------------------------------
extern "C" void kernel_launch(void* const* d_in, const int* in_sizes, int n_in,
                              void* d_out, int out_size, void* d_ws, size_t ws_size,
                              hipStream_t stream)
{
    const float* x  = (const float*)d_in[0];
    const float* wq = (const float*)d_in[1];
    const float* bq = (const float*)d_in[2];
    const float* wk = (const float*)d_in[3];
    const float* bk = (const float*)d_in[4];
    const float* wv = (const float*)d_in[5];
    const float* bv = (const float*)d_in[6];
    const float* wo = (const float*)d_in[7];
    const float* bo = (const float*)d_in[8];

    char* ws = (char*)d_ws;
    size_t off = 0;
    auto suballoc = [&](size_t bytes) -> void* {
        void* p = ws + off;
        off += (bytes + 255) & ~(size_t)255;
        return p;
    };

    const size_t NME = (size_t)B_ * T_ * E_;        // 8388608
    const size_t NW  = (size_t)E_ * E_;             // 1048576

    bf16_t* xbf    = (bf16_t*)suballoc(NME * 2);
    bf16_t* wqt    = (bf16_t*)suballoc(NW * 2);     // transposed [n][k]
    bf16_t* wkt    = (bf16_t*)suballoc(NW * 2);
    bf16_t* wvt    = (bf16_t*)suballoc(NW * 2);
    bf16_t* wot    = (bf16_t*)suballoc(NW * 2);
    bf16_t* qbuf   = (bf16_t*)suballoc(NME * 2);    // [BH,T,D]
    bf16_t* kbuf   = (bf16_t*)suballoc(NME * 2);
    bf16_t* vbuf   = (bf16_t*)suballoc(NME * 2);
    float*  Sbuf   = (float*)suballoc((size_t)BH_ * NCHUNK * 64 * 64 * 4);
    float*  zbuf   = (float*)suballoc((size_t)BH_ * NCHUNK * 64 * 4);
    bf16_t* attnbf = (bf16_t*)suballoc(NME * 2);
    (void)ws_size;

    // 1) conversions / weight transposes
    {
        int n = (int)NME;
        f32_to_bf16_kernel<<<(n + 255) / 256, 256, 0, stream>>>(x, xbf, n);
        dim3 tg(E_ / 32, E_ / 32);   // (32, 32)
        w_transpose_bf16_kernel<<<tg, 256, 0, stream>>>(wq, wqt);
        w_transpose_bf16_kernel<<<tg, 256, 0, stream>>>(wk, wkt);
        w_transpose_bf16_kernel<<<tg, 256, 0, stream>>>(wv, wvt);
        w_transpose_bf16_kernel<<<tg, 256, 0, stream>>>(wo, wot);
    }

    // 2) Q/K/V projections (fused bias + feature map)
    dim3 gemm_grid(E_ / 128, (B_ * T_) / 128);      // (8, 64)
    proj_gemm_kernel<<<gemm_grid, 256, 0, stream>>>(xbf, wqt, bq, qbuf, nullptr, 0, 1);
    proj_gemm_kernel<<<gemm_grid, 256, 0, stream>>>(xbf, wkt, bk, kbuf, nullptr, 0, 1);
    proj_gemm_kernel<<<gemm_grid, 256, 0, stream>>>(xbf, wvt, bv, vbuf, nullptr, 0, 0);

    // 3) per-chunk K^T V + k-sums
    chunk_state_kernel<<<BH_ * NCHUNK, 128, 0, stream>>>(kbuf, vbuf, Sbuf, zbuf);

    // 4) exclusive prefix over chunks
    prefix_kernel<<<BH_, 256, 0, stream>>>(Sbuf, zbuf);

    // 5) per-chunk attention output
    attn_chunk_kernel<<<BH_ * NCHUNK, 128, 0, stream>>>(qbuf, kbuf, vbuf, Sbuf, zbuf, attnbf);

    // 6) final projection -> f32 d_out
    proj_gemm_kernel<<<gemm_grid, 256, 0, stream>>>(attnbf, wot, bo, nullptr,
                                                    (float*)d_out, 1, 0);
}
